// GVAE_6064493822275
// MI455X (gfx1250) — matrix operand order repaired and checked by reading, not compiled, not run on hardware
//
#include <hip/hip_runtime.h>
#include <math.h>

// ---------------------------------------------------------------------------
// Graph-VAE (GCN encoder/decoder) for MI455X / gfx1250.
// Dense GEMMs use v_wmma_f32_16x16x32_bf16 (f32 accumulate); sparse
// aggregation uses streaming float4 + global_atomic_add_f32 (bandwidth-bound).
// ---------------------------------------------------------------------------

typedef __attribute__((ext_vector_type(16))) __bf16 v16bf;
typedef __attribute__((ext_vector_type(8)))  float  v8f;

#define NNODES 30000
#define NEDGES 480000
#define DIN 512
#define HID 256
#define LAT 64

__device__ __forceinline__ unsigned short f32_to_bf16(float f) {
  union { float f; unsigned u; } v; v.f = f;
  unsigned r = v.u + 0x7FFFu + ((v.u >> 16) & 1u);  // round-to-nearest-even
  return (unsigned short)(r >> 16);
}

__global__ void k_cvt_bf16(const float* __restrict__ in,
                           unsigned short* __restrict__ out, int n) {
  int i = blockIdx.x * blockDim.x + threadIdx.x;
  if (i < n) out[i] = f32_to_bf16(in[i]);
}

// W [K,N] f32 row-major -> Wt [N,K] bf16 row-major (B pre-transposed for WMMA)
__global__ void k_transpose_w(const float* __restrict__ W,
                              unsigned short* __restrict__ Wt, int K, int N) {
  int i = blockIdx.x * blockDim.x + threadIdx.x;
  if (i >= N * K) return;
  int n = i / K, k = i % K;
  Wt[i] = f32_to_bf16(W[(size_t)k * N + n]);
}

__global__ void k_deg_init(float* __restrict__ deg, int n) {
  int i = blockIdx.x * blockDim.x + threadIdx.x;
  if (i < n) deg[i] = 1.0f;  // self-loop weight
}

__global__ void k_deg_scatter(const int* __restrict__ dst,
                              const float* __restrict__ w,
                              float* __restrict__ deg, int E) {
  int e = blockIdx.x * blockDim.x + threadIdx.x;
  if (e < E)
    __hip_atomic_fetch_add(&deg[dst[e]], w[e], __ATOMIC_RELAXED,
                           __HIP_MEMORY_SCOPE_AGENT);
}

__global__ void k_dinv(float* __restrict__ deg_dinv,
                       float* __restrict__ selfnorm, int n) {
  int i = blockIdx.x * blockDim.x + threadIdx.x;
  if (i >= n) return;
  float d = deg_dinv[i];
  float di = (d > 0.0f) ? rsqrtf(d) : 0.0f;
  deg_dinv[i] = di;
  selfnorm[i] = di * di;  // dinv[i] * 1.0 * dinv[i]
}

__global__ void k_edge_norm(const int* __restrict__ src,
                            const int* __restrict__ dst,
                            const float* __restrict__ w,
                            const float* __restrict__ dinv,
                            float* __restrict__ norm, int E) {
  int e = blockIdx.x * blockDim.x + threadIdx.x;
  if (e < E) norm[e] = dinv[src[e]] * w[e] * dinv[dst[e]];
}

// ---------------------------------------------------------------------------
// WMMA GEMM: C[M,N] (f32) = A[M,K] (bf16, row-major) x Bt[N,K]^T (bf16).
// One wave -> 16x64 C strip (4 accumulators); K stepped by 32.
// 16-bit A 16x32, B 32x16, f32 C 16x16; C per CDNA5 ISA 7.12.2, A/B in contiguous k order (k = 16h + s).
// ---------------------------------------------------------------------------
__global__ __launch_bounds__(256)
void k_gemm_wmma(const __bf16* __restrict__ A, const __bf16* __restrict__ Bt,
                 float* __restrict__ C, int M, int N, int K) {
  const int lane = threadIdx.x & 31;
  const int wave = threadIdx.x >> 5;
  const int groups_n = N >> 6;                 // 64-wide N groups
  const int total = (M >> 4) * groups_n;
  int g = blockIdx.x * 8 + wave;
  if (g >= total) return;
  const int m0 = (g / groups_n) << 4;
  const int n0 = (g % groups_n) << 6;

  const int half = lane >> 4;                  // 0 | 1
  const int l16  = lane & 15;

  v8f acc0 = {}, acc1 = {}, acc2 = {}, acc3 = {};

  const __bf16* Arow = A + (size_t)(m0 + l16) * K;        // this lane's A row
  const __bf16* B0 = Bt + (size_t)(n0 + 0  + l16) * K + 16 * half;
  const __bf16* B1 = Bt + (size_t)(n0 + 16 + l16) * K + 16 * half;
  const __bf16* B2 = Bt + (size_t)(n0 + 32 + l16) * K + 16 * half;
  const __bf16* B3 = Bt + (size_t)(n0 + 48 + l16) * K + 16 * half;

  for (int k0 = 0; k0 < K; k0 += 32) {
    // A frag: elems 0..15 = A[m][k0+16h .. +15] (one contiguous 32B chunk)
    union { v16bf v; uint4 u[2]; } af;
    af.u[0] = *(const uint4*)(Arow + k0 + 16 * half);
    af.u[1] = *(const uint4*)(Arow + k0 + 16 * half + 8);
    // B frag: elems 0..15 = Bt[n][k0+16h .. +15] (one contiguous 32B chunk)
    union { v16bf v; uint4 u[2]; } b0, b1, b2, b3;
    b0.u[0] = *(const uint4*)(B0 + k0); b0.u[1] = *(const uint4*)(B0 + k0 + 8);
    b1.u[0] = *(const uint4*)(B1 + k0); b1.u[1] = *(const uint4*)(B1 + k0 + 8);
    b2.u[0] = *(const uint4*)(B2 + k0); b2.u[1] = *(const uint4*)(B2 + k0 + 8);
    b3.u[0] = *(const uint4*)(B3 + k0); b3.u[1] = *(const uint4*)(B3 + k0 + 8);

    acc0 = __builtin_amdgcn_wmma_f32_16x16x32_bf16(false, af.v, false, b0.v,
                                                   (short)0, acc0, false, false);
    acc1 = __builtin_amdgcn_wmma_f32_16x16x32_bf16(false, af.v, false, b1.v,
                                                   (short)0, acc1, false, false);
    acc2 = __builtin_amdgcn_wmma_f32_16x16x32_bf16(false, af.v, false, b2.v,
                                                   (short)0, acc2, false, false);
    acc3 = __builtin_amdgcn_wmma_f32_16x16x32_bf16(false, af.v, false, b3.v,
                                                   (short)0, acc3, false, false);
  }

  // C/D layout: lane (l16, half), VGPR r holds row m0 + r + 8*half, col n
  float* Cb = C + (size_t)(m0 + 8 * half) * N + l16;
#pragma unroll
  for (int r = 0; r < 8; ++r) {
    float* Cr = Cb + (size_t)r * N + n0;
    Cr[0]  = acc0[r];
    Cr[16] = acc1[r];
    Cr[32] = acc2[r];
    Cr[48] = acc3[r];
  }
}

// out[i,f] = selfnorm[i] * h2[i,f]   (self-loop term, initializes accumulator)
__global__ void k_self_init(const float* __restrict__ h2,
                            const float* __restrict__ selfnorm,
                            float* __restrict__ out, int F, int total4) {
  int i = blockIdx.x * blockDim.x + threadIdx.x;
  if (i >= total4) return;
  int node = i / (F >> 2);
  float s = selfnorm[node];
  float4 v = ((const float4*)h2)[i];
  v.x *= s; v.y *= s; v.z *= s; v.w *= s;
  ((float4*)out)[i] = v;
}

// out[dst,f] += norm[e] * h2[src,f]   (float4 per thread, native f32 atomics)
__global__ void k_edge_scatter(const float* __restrict__ h2,
                               const int* __restrict__ src,
                               const int* __restrict__ dst,
                               const float* __restrict__ norm,
                               float* __restrict__ out, int F, long total) {
  long i = (long)blockIdx.x * blockDim.x + threadIdx.x;
  if (i >= total) return;
  int c4 = F >> 2;
  int e = (int)(i / c4);
  int c = (int)(i % c4);
  float nv = norm[e];
  int s = src[e], d = dst[e];
  float4 v = ((const float4*)(h2 + (size_t)s * F))[c];
  float* o = out + (size_t)d * F + 4 * c;
  __hip_atomic_fetch_add(o + 0, v.x * nv, __ATOMIC_RELAXED, __HIP_MEMORY_SCOPE_AGENT);
  __hip_atomic_fetch_add(o + 1, v.y * nv, __ATOMIC_RELAXED, __HIP_MEMORY_SCOPE_AGENT);
  __hip_atomic_fetch_add(o + 2, v.z * nv, __ATOMIC_RELAXED, __HIP_MEMORY_SCOPE_AGENT);
  __hip_atomic_fetch_add(o + 3, v.w * nv, __ATOMIC_RELAXED, __HIP_MEMORY_SCOPE_AGENT);
}

// out += bias; optional ReLU; optional bf16 copy for the next layer's WMMA input
__global__ void k_epilogue(float* __restrict__ out, const float* __restrict__ bias,
                           unsigned short* __restrict__ out_bf16, int F, int relu,
                           int total) {
  int i = blockIdx.x * blockDim.x + threadIdx.x;
  if (i >= total) return;
  float v = out[i] + bias[i % F];
  if (relu) v = v > 0.0f ? v : 0.0f;
  out[i] = v;
  if (out_bf16) out_bf16[i] = f32_to_bf16(v);
}

__device__ __forceinline__ unsigned hash_u32(unsigned x) {
  x ^= x >> 17; x *= 0xED5AD4BBu;
  x ^= x >> 11; x *= 0xAC4C1B51u;
  x ^= x >> 15; x *= 0x31848BABu;
  x ^= x >> 14;
  return x;
}

// z = mu + eps * exp(0.5*logvar); deterministic hash+Box-Muller eps
__global__ void k_reparam(const float* __restrict__ mu, const float* __restrict__ lv,
                          float* __restrict__ z, unsigned short* __restrict__ zb,
                          int total) {
  int i = blockIdx.x * blockDim.x + threadIdx.x;
  if (i >= total) return;
  unsigned a = hash_u32(2u * (unsigned)i + 0x9E3779B9u);
  unsigned b = hash_u32(2u * (unsigned)i + 0x85EBCA77u);
  float u1 = (float)(a >> 8) * (1.0f / 16777216.0f) + (0.5f / 16777216.0f);
  float u2 = (float)(b >> 8) * (1.0f / 16777216.0f);
  float r = sqrtf(-2.0f * logf(u1));
  float eps = r * __cosf(6.28318530718f * u2);
  float v = mu[i] + eps * expf(0.5f * lv[i]);
  z[i] = v;
  zb[i] = f32_to_bf16(v);
}

// ---------------------------------------------------------------------------
extern "C" void kernel_launch(void* const* d_in, const int* in_sizes, int n_in,
                              void* d_out, int out_size, void* d_ws, size_t ws_size,
                              hipStream_t stream) {
  (void)in_sizes; (void)n_in; (void)out_size; (void)ws_size;

  const float* x      = (const float*)d_in[0];
  const int*   ei     = (const int*)d_in[1];
  const int*   src    = ei;
  const int*   dst    = ei + NEDGES;
  const float* w      = (const float*)d_in[2];
  const float* W_enc1 = (const float*)d_in[3];  const float* b_enc1 = (const float*)d_in[4];
  const float* W_mu   = (const float*)d_in[5];  const float* b_mu   = (const float*)d_in[6];
  const float* W_lv   = (const float*)d_in[7];  const float* b_lv   = (const float*)d_in[8];
  const float* W_dec1 = (const float*)d_in[9];  const float* b_dec1 = (const float*)d_in[10];
  const float* W_dec2 = (const float*)d_in[11]; const float* b_dec2 = (const float*)d_in[12];

  float* out_recon = (float*)d_out;                                  // [N, DIN]
  float* out_mu    = out_recon + (size_t)NNODES * DIN;               // [N, LAT]
  float* out_lv    = out_mu    + (size_t)NNODES * LAT;               // [N, LAT]

  // ---- workspace bump allocator (256B aligned) ----
  char*  ws  = (char*)d_ws;
  size_t off = 0;
  auto alloc = [&](size_t bytes) -> char* {
    char* p = ws + off;
    off = (off + bytes + 255) & ~(size_t)255;
    return p;
  };
  float*          dinv     = (float*)alloc((size_t)NNODES * 4);   // deg, then dinv (in place)
  float*          selfnorm = (float*)alloc((size_t)NNODES * 4);
  float*          normbuf  = (float*)alloc((size_t)NEDGES * 4);
  unsigned short* x_bf     = (unsigned short*)alloc((size_t)NNODES * DIN * 2);
  float*          h1       = (float*)alloc((size_t)NNODES * HID * 4);
  unsigned short* h1_bf    = (unsigned short*)alloc((size_t)NNODES * HID * 2);
  float*          zf       = (float*)alloc((size_t)NNODES * LAT * 4);
  unsigned short* z_bf     = (unsigned short*)alloc((size_t)NNODES * LAT * 2);
  float*          hd       = (float*)alloc((size_t)NNODES * HID * 4);
  unsigned short* hd_bf    = (unsigned short*)alloc((size_t)NNODES * HID * 2);
  float*          h2       = (float*)alloc((size_t)NNODES * DIN * 4);  // GEMM scratch
  unsigned short* Wt_enc1  = (unsigned short*)alloc((size_t)HID * DIN * 2);
  unsigned short* Wt_mu    = (unsigned short*)alloc((size_t)LAT * HID * 2);
  unsigned short* Wt_lv    = (unsigned short*)alloc((size_t)LAT * HID * 2);
  unsigned short* Wt_dec1  = (unsigned short*)alloc((size_t)HID * LAT * 2);
  unsigned short* Wt_dec2  = (unsigned short*)alloc((size_t)DIN * HID * 2);

  const int T = 256;
  auto blks = [&](long n) { return (unsigned)((n + T - 1) / T); };

  // ---- activation + weight conversion ----
  k_cvt_bf16<<<blks((long)NNODES * DIN), T, 0, stream>>>(x, x_bf, NNODES * DIN);
  k_transpose_w<<<blks((long)HID * DIN), T, 0, stream>>>(W_enc1, Wt_enc1, DIN, HID);
  k_transpose_w<<<blks((long)LAT * HID), T, 0, stream>>>(W_mu,   Wt_mu,   HID, LAT);
  k_transpose_w<<<blks((long)LAT * HID), T, 0, stream>>>(W_lv,   Wt_lv,   HID, LAT);
  k_transpose_w<<<blks((long)HID * LAT), T, 0, stream>>>(W_dec1, Wt_dec1, LAT, HID);
  k_transpose_w<<<blks((long)DIN * HID), T, 0, stream>>>(W_dec2, Wt_dec2, HID, DIN);

  // ---- graph normalization: deg -> dinv -> per-edge norm ----
  k_deg_init<<<blks(NNODES), T, 0, stream>>>(dinv, NNODES);
  k_deg_scatter<<<blks(NEDGES), T, 0, stream>>>(dst, w, dinv, NEDGES);
  k_dinv<<<blks(NNODES), T, 0, stream>>>(dinv, selfnorm, NNODES);
  k_edge_norm<<<blks(NEDGES), T, 0, stream>>>(src, dst, w, dinv, normbuf, NEDGES);

  // ---- one GCN conv: WMMA GEMM -> self init -> edge scatter -> epilogue ----
  auto conv = [&](const unsigned short* Abf, int K, const unsigned short* Wt, int F,
                  const float* bias, bool relu, float* outf, unsigned short* outbf) {
    int groups = (NNODES / 16) * (F / 64);
    k_gemm_wmma<<<(groups + 7) / 8, 256, 0, stream>>>(
        (const __bf16*)Abf, (const __bf16*)Wt, h2, NNODES, F, K);
    int total4 = NNODES * (F / 4);
    k_self_init<<<blks(total4), T, 0, stream>>>(h2, selfnorm, outf, F, total4);
    long totE = (long)NEDGES * (F / 4);
    k_edge_scatter<<<blks(totE), T, 0, stream>>>(h2, src, dst, normbuf, outf, F, totE);
    int tot = NNODES * F;
    k_epilogue<<<blks(tot), T, 0, stream>>>(outf, bias, outbf, F, relu ? 1 : 0, tot);
  };

  // encoder
  conv(x_bf,  DIN, Wt_enc1, HID, b_enc1, true,  h1,     h1_bf);
  conv(h1_bf, HID, Wt_mu,   LAT, b_mu,   false, out_mu, nullptr);
  conv(h1_bf, HID, Wt_lv,   LAT, b_lv,   false, out_lv, nullptr);
  // reparameterize
  k_reparam<<<blks((long)NNODES * LAT), T, 0, stream>>>(out_mu, out_lv, zf, z_bf,
                                                        NNODES * LAT);
  // decoder
  conv(z_bf,  LAT, Wt_dec1, HID, b_dec1, true,  hd,        hd_bf);
  conv(hd_bf, HID, Wt_dec2, DIN, b_dec2, false, out_recon, nullptr);
}
